// FinalTranform_66769561584204
// MI455X (gfx1250) — matrix operand order repaired and checked by reading, not compiled, not run on hardware
//
#include <hip/hip_runtime.h>
#include <hip/hip_bf16.h>
#include <stdint.h>

typedef __attribute__((ext_vector_type(16))) _Float16 v16h;
typedef __attribute__((ext_vector_type(8)))  float    v8f;

// ---------------------------------------------------------------------------
// Output layout (floats):
//   [0, 33226752)            image passthrough (64*3*416*416)
//   [33226752, 33497152)     t13: [64][13][13][25]
//   [33497152, 34578752)     t26: [64][26][26][25]
//   [34578752, 38905152)     t52: [64][52][52][25]
// ---------------------------------------------------------------------------

__device__ __forceinline__ unsigned lds_off(const void* p) {
  // Generic pointer to LDS: low 32 bits are the LDS byte address (ISA 10.2).
  return (unsigned)(uintptr_t)p;
}

// ------------------------- Kernel 1: async image copy ----------------------
// Stream 16B/lane chunks through LDS via the CDNA5 async global<->LDS path.
// 4 slots per thread pipelined between s_wait_asynccnt fences.
__global__ __launch_bounds__(256) void img_copy_async(const float* __restrict__ src,
                                                      float* __restrict__ dst,
                                                      long n16) {
  __shared__ float4 buf[4][256];
  const int tid = threadIdx.x;
  unsigned l0 = lds_off(&buf[0][tid]);
  unsigned l1 = lds_off(&buf[1][tid]);
  unsigned l2 = lds_off(&buf[2][tid]);
  unsigned l3 = lds_off(&buf[3][tid]);
  const long gs = (long)gridDim.x * 256;
  long i = (long)blockIdx.x * 256 + tid;
  const unsigned long long sb = (unsigned long long)(uintptr_t)src;
  const unsigned long long db = (unsigned long long)(uintptr_t)dst;

  while (i + 3 * gs < n16) {
    unsigned long long a0 = sb + (unsigned long long)(i         ) * 16ull;
    unsigned long long a1 = sb + (unsigned long long)(i + gs    ) * 16ull;
    unsigned long long a2 = sb + (unsigned long long)(i + 2 * gs) * 16ull;
    unsigned long long a3 = sb + (unsigned long long)(i + 3 * gs) * 16ull;
    asm volatile("global_load_async_to_lds_b128 %0, %1, off" :: "v"(l0), "v"(a0) : "memory");
    asm volatile("global_load_async_to_lds_b128 %0, %1, off" :: "v"(l1), "v"(a1) : "memory");
    asm volatile("global_load_async_to_lds_b128 %0, %1, off" :: "v"(l2), "v"(a2) : "memory");
    asm volatile("global_load_async_to_lds_b128 %0, %1, off" :: "v"(l3), "v"(a3) : "memory");
    asm volatile("s_wait_asynccnt 0x0" ::: "memory");
    unsigned long long d0 = db + (unsigned long long)(i         ) * 16ull;
    unsigned long long d1 = db + (unsigned long long)(i + gs    ) * 16ull;
    unsigned long long d2 = db + (unsigned long long)(i + 2 * gs) * 16ull;
    unsigned long long d3 = db + (unsigned long long)(i + 3 * gs) * 16ull;
    asm volatile("global_store_async_from_lds_b128 %0, %1, off" :: "v"(d0), "v"(l0) : "memory");
    asm volatile("global_store_async_from_lds_b128 %0, %1, off" :: "v"(d1), "v"(l1) : "memory");
    asm volatile("global_store_async_from_lds_b128 %0, %1, off" :: "v"(d2), "v"(l2) : "memory");
    asm volatile("global_store_async_from_lds_b128 %0, %1, off" :: "v"(d3), "v"(l3) : "memory");
    asm volatile("s_wait_asynccnt 0x0" ::: "memory");  // slots reusable next iter
    i += 4 * gs;
  }
  while (i < n16) {  // tail, one chunk at a time
    unsigned long long a = sb + (unsigned long long)i * 16ull;
    asm volatile("global_load_async_to_lds_b128 %0, %1, off" :: "v"(l0), "v"(a) : "memory");
    asm volatile("s_wait_asynccnt 0x0" ::: "memory");
    unsigned long long d = db + (unsigned long long)i * 16ull;
    asm volatile("global_store_async_from_lds_b128 %0, %1, off" :: "v"(d), "v"(l0) : "memory");
    asm volatile("s_wait_asynccnt 0x0" ::: "memory");
    i += gs;
  }
}

// ------------------- Kernel 2: FCOS targets (cls via WMMA) -----------------
__global__ __launch_bounds__(256) void fcos_targets(const float* __restrict__ bb,
                                                    const int* __restrict__ lab,
                                                    float* __restrict__ out) {
  const int b  = blockIdx.x;
  const int sc = blockIdx.y;  // 0:13, 1:26, 2:52
  int S; float strideF; long base;
  if (sc == 0)      { S = 13; strideF = 32.f; base = 33226752L; }
  else if (sc == 1) { S = 26; strideF = 16.f; base = 33497152L; }
  else              { S = 52; strideF =  8.f; base = 34578752L; }
  const float inv = 1.0f / strideF;  // exact (power of two)
  float* tb = out + base + (long)b * (long)S * S * 25;

  __shared__ float cxs[32], cys[32], hws[32], hhs[32];
  __shared__ float ilo[32], ihi[32], jlo[32], jhi[32];
  __shared__ int   lbs[32];
  const int tid = threadIdx.x;
  if (tid < 32) {
    const float* p = bb + ((long)b * 32 + tid) * 4;
    float cx = p[0], cy = p[1], w = p[2], h = p[3];
    cxs[tid] = cx; cys[tid] = cy;
    hws[tid] = floorf(w * 0.5f); hhs[tid] = floorf(h * 0.5f);
    float p0 = floorf(cx / strideF), p1 = floorf(cy / strideF);
    float b0 = floorf(0.5f * w / strideF), b1 = floorf(0.5f * h / strideF);
    ilo[tid] = p0 - b0; ihi[tid] = p0 + b0;
    jlo[tid] = p1 - b1; jhi[tid] = p1 + b1;
    lbs[tid] = lab[(long)b * 32 + tid];
  }
  __syncthreads();

  // ---- Phase 1: class planes. D = onehot(16x32) x region(32x16) per tile.
  const int lane = tid & 31, wv = tid >> 5;
  const int ll = lane & 15, g = lane >> 4;
  // A-matrix (16-bit 16x32 layout, ISA 7.12.2): lane row M=ll;
  // element e -> K = e+8g (e<8) else e+8+8g.
  v16h a0, a1;
#pragma unroll
  for (int e = 0; e < 16; ++e) {
    int n  = (e < 8) ? (e + 8 * g) : (e + 8 + 8 * g);
    int lb = lbs[n];
    a0[e] = (lb == ll)      ? (_Float16)1.0f : (_Float16)0.0f;
    a1[e] = (lb == ll + 16) ? (_Float16)1.0f : (_Float16)0.0f;
  }
  const int cells = S * S;
  const int nt = (cells + 15) >> 4;
  const v8f cz = {};
  for (int t = wv; t < nt; t += 8) {
    int  cell  = t * 16 + ll;          // B col N = ll
    bool valid = cell < cells;
    int  cc = valid ? cell : (cells - 1);
    int  ii = cc % S, jj = cc / S;
    float fi = (float)ii, fj = (float)jj;
    // B-matrix (32x16): lane col N=ll, element e -> K = e+8g (e<8) else e+8+8g (box index).
    v16h bm;
#pragma unroll
    for (int e = 0; e < 16; ++e) {
      int n = (e < 8) ? (e + 8 * g) : (e + 8 + 8 * g);
      bool r = (fi >= ilo[n]) && (fi < ihi[n]) && (fj >= jlo[n]) && (fj < jhi[n]);
      bm[e] = r ? (_Float16)1.0f : (_Float16)0.0f;
    }
    v8f d0 = __builtin_amdgcn_wmma_f32_16x16x32_f16(false, a0, false, bm, (short)0, cz, false, false);
    v8f d1 = __builtin_amdgcn_wmma_f32_16x16x32_f16(false, a1, false, bm, (short)0, cz, false, false);
    if (valid) {
      float* o = tb + (long)cell * 25;
#pragma unroll
      for (int r = 0; r < 8; ++r)        // D row = r + 8g -> classes 0..15
        o[8 * g + r] = (d0[r] > 0.5f) ? 1.0f : 0.0f;
      if (g == 0) {                      // tile-1 rows 0..3 -> classes 16..19
#pragma unroll
        for (int r = 0; r < 4; ++r)
          o[16 + r] = (d1[r] > 0.5f) ? 1.0f : 0.0f;
      }
    }
  }

  // ---- Phase 2: regression channels, sequential scan over boxes per cell.
  for (int cell = tid; cell < cells; cell += 256) {
    int ii = cell % S, jj = cell / S;
    float fi = (float)ii, fj = (float)jj;
    float ci = (fi + 0.5f) * strideF, cj = (fj + 0.5f) * strideF;
    float s0 = 0.f, s1 = 0.f, s2 = 0.f, s3 = 0.f, s4 = 0.f, regmax = 0.f;
    for (int n = 0; n < 32; ++n) {
      bool region = (fi >= ilo[n]) && (fi < ihi[n]) && (fj >= jlo[n]) && (fj < jhi[n]);
      float L  = ci - (cxs[n] - hws[n]);
      float R  = (cxs[n] + hws[n]) - ci;
      float T  = cj - (cys[n] - hhs[n]);
      float Bo = (cys[n] + hhs[n]) - cj;
      float m  = fmaxf(fmaxf(L, R), fmaxf(T, Bo));
      bool band = (sc == 0) ? (m > 256.f)
                : (sc == 1) ? ((m > 64.f) && (m <= 256.f))
                            : (m <= 64.f);
      if (region && band && (regmax == 0.f || m < regmax)) {
        float cent = __builtin_sqrtf(fminf(L, R) * fminf(T, Bo) /
                                     (fmaxf(L, R) * fmaxf(T, Bo)));
        s0 = fmaxf(L * inv, 0.f);  s1 = fmaxf(T * inv, 0.f);
        s2 = fmaxf(R * inv, 0.f);  s3 = fmaxf(Bo * inv, 0.f);
        s4 = fmaxf(cent, 0.f);
        regmax = fmaxf(fmaxf(fmaxf(s0, s1), fmaxf(s2, s3)), s4);  // reg.max(axis=0)
      }
    }
    float* o = tb + (long)cell * 25 + 20;
    o[0] = s0; o[1] = s1; o[2] = s2; o[3] = s3; o[4] = s4;
  }
}

// ---------------------------------------------------------------------------
extern "C" void kernel_launch(void* const* d_in, const int* in_sizes, int n_in,
                              void* d_out, int out_size, void* d_ws, size_t ws_size,
                              hipStream_t stream) {
  (void)n_in; (void)out_size; (void)d_ws; (void)ws_size;
  const float* img = (const float*)d_in[0];
  const float* bb  = (const float*)d_in[1];
  const int*   lb  = (const int*)d_in[2];
  float* out = (float*)d_out;

  long n16 = (long)in_sizes[0] / 4;  // number of 16-byte chunks in the image
  img_copy_async<<<dim3(1024), dim3(256), 0, stream>>>(img, out, n16);
  fcos_targets<<<dim3(64, 3), dim3(256), 0, stream>>>(bb, lb, out);
}
